// MoE_592705487075
// MI455X (gfx1250) — matrix operand order repaired and checked by reading, not compiled, not run on hardware
//
#include <hip/hip_runtime.h>
#include <hip/hip_bf16.h>

// ---------------------------------------------------------------------------
// MoE top-2 routed kernel for gfx1250 (MI455X).
// bf16x2 split-precision WMMA; all f32->bf16 conversion hoisted into one-shot
// pre-passes; GEMM tile = 32 tokens x 128 outputs so every B (W) fetch is
// reused by two accumulators (halves L2 traffic vs 16-row tiles).
//
// Workspace layout (requires ws_size >= ~102 MB):
//   counts   8 x i32
//   route    [16384] {i0,i1,w0,w1}
//   entries  [8][16384] {tok,w}
//   x_hi/x_lo  bf16 planes [16384][1024]                      (2 x 32 MB)
//   w_hi/w_lo  swizzled bf16 planes [8][32][1024][16 dwords]  (2 x 16 MB)
// ---------------------------------------------------------------------------

#define TOKENS      16384
#define NUM_EXPERTS 8
#define DIM         1024

typedef __bf16    v16bf __attribute__((ext_vector_type(16)));
typedef float     v8f   __attribute__((ext_vector_type(8)));
typedef unsigned  u32x4 __attribute__((ext_vector_type(4)));
typedef int       i32x4 __attribute__((ext_vector_type(4)));
typedef int       i32x8 __attribute__((ext_vector_type(8)));

struct Route { int i0, i1; float w0, w1; };
struct Entry { int tok; float w; };

union BfPack {
    v16bf v;
    uint4 q[2];
};

#if __has_builtin(__builtin_amdgcn_tensor_load_to_lds) && \
    __has_builtin(__builtin_amdgcn_s_wait_tensorcnt)
#define HAVE_TDM 1
#else
#define HAVE_TDM 0
#endif

__device__ __forceinline__ unsigned short f32_to_bf16_rne(float f) {
    unsigned u = __builtin_bit_cast(unsigned, f);
    unsigned r = u + 0x7FFFu + ((u >> 16) & 1u);
    return (unsigned short)(r >> 16);
}
__device__ __forceinline__ float bf16_bits_to_f32(unsigned short h) {
    unsigned u = ((unsigned)h) << 16;
    return __builtin_bit_cast(float, u);
}
__device__ __forceinline__ void split_bf16(float f, unsigned short& hi, unsigned short& lo) {
    hi = f32_to_bf16_rne(f);
    lo = f32_to_bf16_rne(f - bf16_bits_to_f32(hi));
}

// ---------------------------------------------------------------------------
// Kernel 0: zero routing counters
// ---------------------------------------------------------------------------
__global__ void zero_counts_kernel(int* counts) {
    if (threadIdx.x < NUM_EXPERTS) counts[threadIdx.x] = 0;
}

// ---------------------------------------------------------------------------
// Kernel 1: split x into bf16 hi/lo planes (packed pairs as dwords)
// ---------------------------------------------------------------------------
__global__ __launch_bounds__(256) void split_x_kernel(
    const float4* __restrict__ x4, uint2* __restrict__ xhi, uint2* __restrict__ xlo) {
    const int i = blockIdx.x * blockDim.x + threadIdx.x;  // over TOKENS*DIM/4
    const float4 v = x4[i];
    unsigned short h0, l0, h1, l1, h2, l2, h3, l3;
    split_bf16(v.x, h0, l0); split_bf16(v.y, h1, l1);
    split_bf16(v.z, h2, l2); split_bf16(v.w, h3, l3);
    uint2 ph, pl;
    ph.x = (unsigned)h0 | ((unsigned)h1 << 16);
    ph.y = (unsigned)h2 | ((unsigned)h3 << 16);
    pl.x = (unsigned)l0 | ((unsigned)l1 << 16);
    pl.y = (unsigned)l2 | ((unsigned)l3 << 16);
    xhi[i] = ph; xlo[i] = pl;
}

// ---------------------------------------------------------------------------
// Kernel 2: split + swizzle W_experts into the WMMA B-operand layout.
// Unit per (e, kb, n): 16 dwords = ascending K-pairs of the 32-wide K block.
// A lane (col=n, half=h) reads one contiguous 32B run at byte offset h*32.
// grid = (DIM/256, 32, NUM_EXPERTS), block = 256
// ---------------------------------------------------------------------------
__global__ __launch_bounds__(256) void swizzle_w_kernel(
    const float* __restrict__ W, unsigned* __restrict__ whi, unsigned* __restrict__ wlo) {
    const int e  = blockIdx.z;
    const int kb = blockIdx.y;
    const int nt = blockIdx.x;
    const int t  = threadIdx.x;
    __shared__ unsigned sh[256 * 16];  // 16 KB
    __shared__ unsigned sl[256 * 16];  // 16 KB

    const int n = nt * 256 + t;
    const float* Wb = W + ((size_t)e * DIM + (size_t)kb * 32) * DIM + n;
#pragma unroll
    for (int p = 0; p < 16; ++p) {
        const float f0 = Wb[(size_t)(2 * p) * DIM];
        const float f1 = Wb[(size_t)(2 * p + 1) * DIM];
        unsigned short h0, l0, h1, l1;
        split_bf16(f0, h0, l0);
        split_bf16(f1, h1, l1);
        sh[t * 16 + p] = (unsigned)h0 | ((unsigned)h1 << 16);
        sl[t * 16 + p] = (unsigned)l0 | ((unsigned)l1 << 16);
    }
    __syncthreads();
    const size_t base = (((size_t)e * 32 + kb) * DIM + (size_t)nt * 256) * 16; // dwords
    uint4* dh = (uint4*)(whi + base);
    uint4* dl = (uint4*)(wlo + base);
    const uint4* s4h = (const uint4*)sh;
    const uint4* s4l = (const uint4*)sl;
#pragma unroll
    for (int i = 0; i < 4; ++i) {
        dh[i * 256 + t] = s4h[i * 256 + t];
        dl[i * 256 + t] = s4l[i * 256 + t];
    }
}

// ---------------------------------------------------------------------------
// Kernel 3: gating — one wave32 per token, top-2, compacted expert lists
// ---------------------------------------------------------------------------
__global__ __launch_bounds__(256) void gate_kernel(
    const float* __restrict__ x, const float* __restrict__ Wg,
    const float* __restrict__ bg, Route* __restrict__ route,
    int* __restrict__ counts, Entry* __restrict__ entries) {

    const int wave = threadIdx.x >> 5;
    const int lane = threadIdx.x & 31;
    const int t    = blockIdx.x * (blockDim.x >> 5) + wave;

    float acc[NUM_EXPERTS];
#pragma unroll
    for (int e = 0; e < NUM_EXPERTS; ++e) acc[e] = 0.f;

    const float* xr = x + (size_t)t * DIM;
    for (int d = lane; d < DIM; d += 32) {
        const float xv = xr[d];
        const float* wrow = Wg + d * NUM_EXPERTS;
#pragma unroll
        for (int e = 0; e < NUM_EXPERTS; ++e) acc[e] += xv * wrow[e];
    }
#pragma unroll
    for (int e = 0; e < NUM_EXPERTS; ++e) {
#pragma unroll
        for (int off = 16; off > 0; off >>= 1)
            acc[e] += __shfl_xor(acc[e], off, 32);
    }

    if (lane == 0) {
        float l[NUM_EXPERTS];
#pragma unroll
        for (int e = 0; e < NUM_EXPERTS; ++e) l[e] = acc[e] + bg[e];
        int i0 = 0;
#pragma unroll
        for (int e = 1; e < NUM_EXPERTS; ++e) if (l[e] > l[i0]) i0 = e;
        int i1 = (i0 == 0) ? 1 : 0;
#pragma unroll
        for (int e = 0; e < NUM_EXPERTS; ++e)
            if (e != i0 && l[e] > l[i1]) i1 = e;
        const float w0 = 1.f / (1.f + __expf(l[i1] - l[i0]));
        const float w1 = 1.f - w0;
        Route r; r.i0 = i0; r.i1 = i1; r.w0 = w0; r.w1 = w1;
        route[t] = r;
        int p0 = atomicAdd(&counts[i0], 1);
        entries[i0 * TOKENS + p0] = Entry{t, w0};
        int p1 = atomicAdd(&counts[i1], 1);
        entries[i1 * TOKENS + p1] = Entry{t, w1};
    }
}

// ---------------------------------------------------------------------------
// Kernel 4: out = w0*b[i0] + w1*b[i1]  (initializes every output element)
// ---------------------------------------------------------------------------
__global__ __launch_bounds__(256) void init_out_kernel(
    const float4* __restrict__ be4, const Route* __restrict__ route,
    float4* __restrict__ out4) {
    const int idx = blockIdx.x * blockDim.x + threadIdx.x;
    const int t = idx >> 8;
    const int c = idx & 255;
    const Route r = route[t];
    const float4 b0 = be4[r.i0 * (DIM / 4) + c];
    const float4 b1 = be4[r.i1 * (DIM / 4) + c];
    float4 o;
    o.x = r.w0 * b0.x + r.w1 * b1.x;
    o.y = r.w0 * b0.y + r.w1 * b1.y;
    o.z = r.w0 * b0.z + r.w1 * b1.z;
    o.w = r.w0 * b0.w + r.w1 * b1.w;
    out4[idx] = o;
}

// ---------------------------------------------------------------------------
// Kernel 5: routed expert GEMM. Block = 32 tokens x 128 outputs, 8 waves.
// Each wave holds two 16x16 f32 accumulators (token rows 0-15 and 16-31 of
// the tile) so every B fetch is used twice. A panels gathered into LDS via
// TDM gather-mode (16 row indices per descriptor, 4 descriptors) when
// available. Inner loop: 8x ds_load_b128 + 4x global_load_b128 + 6x wmma.
// ---------------------------------------------------------------------------
__global__ __launch_bounds__(256, 2) void expert_gemm_kernel(
    const unsigned short* __restrict__ xhi, const unsigned short* __restrict__ xlo,
    const unsigned* __restrict__ whi, const unsigned* __restrict__ wlo,
    const int* __restrict__ counts, const Entry* __restrict__ entries,
    float* __restrict__ out) {

    const int e   = blockIdx.z;
    const int cnt = counts[e];
    const int m0  = blockIdx.y * 32;
    if (m0 >= cnt) return;  // uniform across block

    __shared__ __align__(16) unsigned short a_hi[32][DIM];  // 64 KB
    __shared__ __align__(16) unsigned short a_lo[32][DIM];  // 64 KB
    __shared__ int   s_tok[32];
    __shared__ float s_w[32];

    const int tid = threadIdx.x;
    if (tid < 32) {
        const int r = m0 + tid;
        if (r < cnt) {
            Entry en = entries[(size_t)e * TOKENS + r];
            s_tok[tid] = en.tok; s_w[tid] = en.w;
        } else {
            s_tok[tid] = 0; s_w[tid] = 0.f;
        }
    }

#if HAVE_TDM
    // ---- TDM gather: 2 x 16 token rows -> LDS, hi & lo planes (4 ops) ----
    if (tid < 32) {
        // D# group 1 (ISA 8.4): data_size=2B, tensor 1024x16384, tile row
        // width 1024 elems, 16 gather indices, dim0 stride 1024 elems.
        i32x8 g1;
        g1[0] = 0x00010000;                 // wg_mask=0, data_size=1 (2B)
        g1[1] = (int)(1024u << 16);         // tensor_dim0 = 1024 (bits 79:48)
        g1[2] = (int)(16384u << 16);        // tensor_dim1 = 16384 (bits 111:80)
        g1[3] = (int)(1024u << 16);         // tile_dim0 = 1024 (bits 127:112)
        g1[4] = 16;                         // tile_dim1 = 16 valid indices
        g1[5] = 1024;                       // tensor_dim0_stride (low 32)
        g1[6] = 0;
        g1[7] = 0;

#pragma unroll
        for (int hb = 0; hb < 2; ++hb) {    // row blocks 0-15 / 16-31
            unsigned pk[8];
#pragma unroll
            for (int i = 0; i < 8; ++i) {
                const int r0 = m0 + hb * 16 + 2 * i;
                const int r1 = r0 + 1;
                unsigned t0 = (r0 < cnt) ? (unsigned)entries[(size_t)e * TOKENS + r0].tok : 0u;
                unsigned t1 = (r1 < cnt) ? (unsigned)entries[(size_t)e * TOKENS + r1].tok : 0u;
                pk[i] = (unsigned)__builtin_amdgcn_readfirstlane(
                            (int)((t0 & 0xFFFFu) | (t1 << 16)));
            }
            i32x4 g2 = {(int)pk[0], (int)pk[1], (int)pk[2], (int)pk[3]};
            i32x4 g3 = {(int)pk[4], (int)pk[5], (int)pk[6], (int)pk[7]};

#pragma unroll
            for (int plane = 0; plane < 2; ++plane) {
                const unsigned short* src = plane ? xlo : xhi;
                unsigned lds_base = (unsigned)(uintptr_t)(plane ? &a_lo[hb * 16][0]
                                                                : &a_hi[hb * 16][0]);
                unsigned long long ga = (unsigned long long)(uintptr_t)src;
                // D# group 0 (ISA 8.3): count=1, gather_mode=1, 16b idx, type=2
                u32x4 g0;
                g0[0] = 0x80000001u;
                g0[1] = lds_base;
                g0[2] = (unsigned)(ga & 0xFFFFFFFFu);
                g0[3] = (unsigned)((ga >> 32) & 0x01FFFFFFu) | (2u << 30);
#if defined(__clang_major__) && (__clang_major__ >= 23)
                i32x8 z8 = {0, 0, 0, 0, 0, 0, 0, 0};
                __builtin_amdgcn_tensor_load_to_lds(g0, g1, g2, g3, z8, 0);
#else
                __builtin_amdgcn_tensor_load_to_lds(g0, g1, g2, g3, 0);
#endif
            }
        }
        __builtin_amdgcn_s_wait_tensorcnt(0);
    }
    __syncthreads();
#else
    // ---- Fallback: cooperative b128 gather-copy of the bf16 panels ----
    __syncthreads();
    {
        const int row = tid >> 3;       // 32 rows, 8 threads per row
        const int c8  = tid & 7;
        const uint4* srch = (const uint4*)(xhi + (size_t)s_tok[row] * DIM);
        const uint4* srcl = (const uint4*)(xlo + (size_t)s_tok[row] * DIM);
        uint4* dsth = (uint4*)&a_hi[row][0];
        uint4* dstl = (uint4*)&a_lo[row][0];
#pragma unroll 2
        for (int c = c8; c < (DIM * 2) / 16; c += 8) {
            dsth[c] = srch[c];
            dstl[c] = srcl[c];
        }
    }
    __syncthreads();
#endif

    const int wave = tid >> 5;
    const int lane = tid & 31;
    const int col  = lane & 15;
    const int h    = lane >> 4;
    const int n0c  = blockIdx.x * 128 + wave * 16 + col;

    // A operand (contiguous k = 16h + s, same order as B): u[0..3] @ row*2048 + h*32,
    //                                    u[4..7] @ row*2048 + 16 + h*32
    const char* aH0 = (const char*)a_hi + col * (DIM * 2) + h * 32;
    const char* aL0 = (const char*)a_lo + col * (DIM * 2) + h * 32;
    const char* aH1 = aH0 + 16 * (DIM * 2);
    const char* aL1 = aL0 + 16 * (DIM * 2);
    // B operand (contiguous k = 16h + s, same order as A): contiguous 32B run at h*32 within
    // the 64B swizzled unit of (e, kb, n0c).
    const char* bHp = (const char*)whi + (((size_t)e * 32) * DIM + n0c) * 64 + h * 32;
    const char* bLp = (const char*)wlo + (((size_t)e * 32) * DIM + n0c) * 64 + h * 32;

    v8f acc0 = {0.f, 0.f, 0.f, 0.f, 0.f, 0.f, 0.f, 0.f};
    v8f acc1 = {0.f, 0.f, 0.f, 0.f, 0.f, 0.f, 0.f, 0.f};

    for (int kb = 0; kb < 32; ++kb) {
        if (kb + 1 < 32) {
            __builtin_prefetch(bHp + (size_t)(kb + 1) * (DIM * 64), 0, 1);
            __builtin_prefetch(bLp + (size_t)(kb + 1) * (DIM * 64), 0, 1);
        }
        BfPack ah0, al0, ah1, al1, bh, bl;
        bh.q[0] = *(const uint4*)(bHp + (size_t)kb * (DIM * 64));
        bh.q[1] = *(const uint4*)(bHp + (size_t)kb * (DIM * 64) + 16);
        bl.q[0] = *(const uint4*)(bLp + (size_t)kb * (DIM * 64));
        bl.q[1] = *(const uint4*)(bLp + (size_t)kb * (DIM * 64) + 16);
        ah0.q[0] = *(const uint4*)(aH0 + kb * 64);
        ah0.q[1] = *(const uint4*)(aH0 + kb * 64 + 16);
        al0.q[0] = *(const uint4*)(aL0 + kb * 64);
        al0.q[1] = *(const uint4*)(aL0 + kb * 64 + 16);
        ah1.q[0] = *(const uint4*)(aH1 + kb * 64);
        ah1.q[1] = *(const uint4*)(aH1 + kb * 64 + 16);
        al1.q[0] = *(const uint4*)(aL1 + kb * 64);
        al1.q[1] = *(const uint4*)(aL1 + kb * 64 + 16);

        acc0 = __builtin_amdgcn_wmma_f32_16x16x32_bf16(
                   false, ah0.v, false, bh.v, (short)0, acc0, false, false);
        acc0 = __builtin_amdgcn_wmma_f32_16x16x32_bf16(
                   false, ah0.v, false, bl.v, (short)0, acc0, false, false);
        acc0 = __builtin_amdgcn_wmma_f32_16x16x32_bf16(
                   false, al0.v, false, bh.v, (short)0, acc0, false, false);
        acc1 = __builtin_amdgcn_wmma_f32_16x16x32_bf16(
                   false, ah1.v, false, bh.v, (short)0, acc1, false, false);
        acc1 = __builtin_amdgcn_wmma_f32_16x16x32_bf16(
                   false, ah1.v, false, bl.v, (short)0, acc1, false, false);
        acc1 = __builtin_amdgcn_wmma_f32_16x16x32_bf16(
                   false, al1.v, false, bh.v, (short)0, acc1, false, false);
    }

    // Epilogue: gate-weighted scatter-accumulate (L2-resident f32 atomics).
#pragma unroll
    for (int v = 0; v < 8; ++v) {
        const int m = v + 8 * h;
        if (m0 + m < cnt) {
            unsafeAtomicAdd(&out[(size_t)s_tok[m] * DIM + n0c], s_w[m] * acc0[v]);
        }
        if (m0 + 16 + m < cnt) {
            unsafeAtomicAdd(&out[(size_t)s_tok[16 + m] * DIM + n0c], s_w[16 + m] * acc1[v]);
        }
    }
}

// ---------------------------------------------------------------------------
// Host-side launcher
// ---------------------------------------------------------------------------
extern "C" void kernel_launch(void* const* d_in, const int* in_sizes, int n_in,
                              void* d_out, int out_size, void* d_ws, size_t ws_size,
                              hipStream_t stream) {
    const float* x  = (const float*)d_in[0];
    const float* We = (const float*)d_in[1];
    const float* be = (const float*)d_in[2];
    const float* Wg = (const float*)d_in[3];
    const float* bg = (const float*)d_in[4];
    float* out = (float*)d_out;

    char* ws = (char*)d_ws;
    const size_t OFF_ROUTE   = 256;
    const size_t OFF_ENTRIES = OFF_ROUTE + (size_t)TOKENS * sizeof(Route);
    const size_t OFF_XHI     = OFF_ENTRIES + (size_t)NUM_EXPERTS * TOKENS * sizeof(Entry);
    const size_t OFF_XLO     = OFF_XHI + (size_t)TOKENS * DIM * 2;
    const size_t OFF_WHI     = OFF_XLO + (size_t)TOKENS * DIM * 2;
    const size_t OFF_WLO     = OFF_WHI + (size_t)NUM_EXPERTS * DIM * DIM * 2;

    int*            counts  = (int*)ws;
    Route*          route   = (Route*)(ws + OFF_ROUTE);
    Entry*          entries = (Entry*)(ws + OFF_ENTRIES);
    unsigned short* xhi     = (unsigned short*)(ws + OFF_XHI);
    unsigned short* xlo     = (unsigned short*)(ws + OFF_XLO);
    unsigned*       whi     = (unsigned*)(ws + OFF_WHI);
    unsigned*       wlo     = (unsigned*)(ws + OFF_WLO);

    zero_counts_kernel<<<dim3(1), dim3(32), 0, stream>>>(counts);

    gate_kernel<<<dim3(TOKENS / 8), dim3(256), 0, stream>>>(
        x, Wg, bg, route, counts, entries);

    split_x_kernel<<<dim3((TOKENS * DIM / 4) / 256), dim3(256), 0, stream>>>(
        (const float4*)x, (uint2*)xhi, (uint2*)xlo);

    swizzle_w_kernel<<<dim3(DIM / 256, 32, NUM_EXPERTS), dim3(256), 0, stream>>>(
        We, whi, wlo);

    init_out_kernel<<<dim3(TOKENS), dim3(256), 0, stream>>>(
        (const float4*)be, route, (float4*)out);

    expert_gemm_kernel<<<dim3(DIM / 128, TOKENS / 32, NUM_EXPERTS),
                         dim3(256), 0, stream>>>(
        xhi, xlo, whi, wlo, counts, entries, out);
}
